// LiquidTimeConstantNetwork_69458211111382
// MI455X (gfx1250) — matrix operand order repaired and checked by reading, not compiled, not run on hardware
//
#include <hip/hip_runtime.h>
#include <hip/hip_bf16.h>

typedef __attribute__((ext_vector_type(16))) __bf16 v16bf;
typedef __attribute__((ext_vector_type(8)))  float  v8f;

#define T_STEPS 512
#define BATCH   512
#define DIN     128
#define HID     1024
#define NG      4096      // 4*HID gate outputs
#define KTOT    1152      // DIN + HID
#define KT_FR   36        // KTOT / 32
#define WG_M    64
#define WG_N    128

// Packed B-fragment index: fragment (nt, kt) = 32 lanes x 16 bf16 contiguous.
// Lane = (k_in_tile>=16)*16 + (n&15); element j = k_in_tile & 15.
__device__ __host__ __forceinline__ size_t wpack_idx(int n, int k) {
    int nt = n >> 4, l16 = n & 15;
    int kt = k >> 5, kin = k & 31;
    int lane = ((kin >> 4) << 4) | l16;
    int j = kin & 15;
    return ((size_t)(nt * KT_FR + kt) * 32 + lane) * 16 + j;
}

// ---------------- weight pack: Wcomb = W_ih @ W_in folded, gate-interleaved, bf16 ----------
__global__ void ltc_pack(const float* __restrict__ W_in, const float* __restrict__ b_in,
                         const float* __restrict__ W_ih, const float* __restrict__ W_hh,
                         const float* __restrict__ b_ih, const float* __restrict__ b_hh,
                         __bf16* __restrict__ Wpack, float* __restrict__ bx,
                         float* __restrict__ bsum) {
    const int n = blockIdx.x;                 // permuted gate-output index
    const int r = (n & 3) * HID + (n >> 2);   // original row in 4H
    const int tid = threadIdx.x;              // 128 threads
    const float* wih_row = W_ih + (size_t)r * HID;

    // X part: Wcomb[n][d], one d per thread
    {
        float acc = 0.f;
        const int d = tid;
        for (int hh = 0; hh < HID; ++hh)
            acc += wih_row[hh] * W_in[(size_t)hh * DIN + d];
        Wpack[wpack_idx(n, d)] = (__bf16)acc;
    }
    // H part: copy W_hh row (bf16)
    const float* whh_row = W_hh + (size_t)r * HID;
    for (int idx = tid; idx < HID; idx += 128)
        Wpack[wpack_idx(n, DIN + idx)] = (__bf16)whh_row[idx];

    // folded biases: bx[n] = W_ih[r,:] . b_in ; bsum[n] = b_ih[r] + b_hh[r]
    __shared__ float red[128];
    float p = 0.f;
    for (int hh = tid; hh < HID; hh += 128) p += wih_row[hh] * b_in[hh];
    red[tid] = p; __syncthreads();
    for (int s = 64; s > 0; s >>= 1) { if (tid < s) red[tid] += red[tid + s]; __syncthreads(); }
    if (tid == 0) { bx[n] = red[0]; bsum[n] = b_ih[r] + b_hh[r]; }
}

// ---------------- zero init h, c ----------------
__global__ void ltc_init(float* __restrict__ hbuf, float* __restrict__ cbuf) {
    size_t i = (size_t)blockIdx.x * blockDim.x + threadIdx.x;
    if (i < (size_t)BATCH * HID) { hbuf[i] = 0.f; cbuf[i] = 0.f; }
}

// ---------------- one recurrent step: fused GEMM (WMMA bf16) + LSTM pointwise ----------------
__global__ void __launch_bounds__(256)
ltc_step(const float* __restrict__ seq, const float* __restrict__ tiv,
         const __bf16* __restrict__ Wpack, const float* __restrict__ bx,
         const float* __restrict__ bsum, float* __restrict__ hbuf,
         float* __restrict__ cbuf, int t) {
    __shared__ __align__(16) __bf16 As[2][WG_M][48];   // A staging (pad 32->48 for banks)
    __shared__ float Gs[WG_M][WG_N];                   // gate tile for fused pointwise

    const int tid   = threadIdx.x;
    const int lane  = tid & 31;
    const int wave  = tid >> 5;          // 8 waves
    const int waveM = wave & 1;          // 2 x M
    const int waveN = wave >> 1;         // 4 x N
    const int m0g   = blockIdx.x * WG_M; // batch rows
    const int n0g   = blockIdx.y * WG_N; // gate cols (interleaved)

    const float dt = tiv[t] * (1.0f / 60.0f);
    const float tw = fminf(fmaxf(dt, 0.1f), 10.0f);

    v8f acc[2][2];
    #pragma unroll
    for (int mf = 0; mf < 2; ++mf)
        #pragma unroll
        for (int nf = 0; nf < 2; ++nf) acc[mf][nf] = {};

    // cooperative A-tile fill: f32 -> bf16, tw-scaled seq slab for k<128, h for k>=128
    const int arow = tid >> 2;            // 0..63
    const int akk  = (tid & 3) * 8;       // 0,8,16,24
    auto fillA = [&](int buf, int kt) {
        const int kg = kt * 32 + akk;
        float4 x0, x1;
        if (kg < DIN) {
            const float* src = seq + ((size_t)t * BATCH + (m0g + arow)) * DIN + kg;
            x0 = *(const float4*)(src);
            x1 = *(const float4*)(src + 4);
            x0.x *= tw; x0.y *= tw; x0.z *= tw; x0.w *= tw;
            x1.x *= tw; x1.y *= tw; x1.z *= tw; x1.w *= tw;
        } else {
            const float* src = hbuf + (size_t)(m0g + arow) * HID + (kg - DIN);
            x0 = *(const float4*)(src);
            x1 = *(const float4*)(src + 4);
        }
        union { __bf16 b[8]; uint4 q; } u;
        u.b[0] = (__bf16)x0.x; u.b[1] = (__bf16)x0.y; u.b[2] = (__bf16)x0.z; u.b[3] = (__bf16)x0.w;
        u.b[4] = (__bf16)x1.x; u.b[5] = (__bf16)x1.y; u.b[6] = (__bf16)x1.z; u.b[7] = (__bf16)x1.w;
        *(uint4*)&As[buf][arow][akk] = u.q;
    };

    fillA(0, 0);
    __syncthreads();

    const int klo = (lane < 16) ? 0 : 16; // A-fragment K gather base per half-wave
    int buf = 0;
    for (int kt = 0; kt < KT_FR; ++kt) {
        if (kt + 1 < KT_FR) fillA(buf ^ 1, kt + 1);

        // B fragments: pre-packed, one 32B/lane vector load each
        v16bf bfr[2];
        #pragma unroll
        for (int nf = 0; nf < 2; ++nf) {
            const int nt = blockIdx.y * 8 + waveN * 2 + nf;
            bfr[nf] = *(const v16bf*)(Wpack + ((size_t)(nt * KT_FR + kt) * 32 + lane) * 16);
        }
        #pragma unroll
        for (int mf = 0; mf < 2; ++mf) {
            const int row = waveM * 32 + mf * 16 + (lane & 15);
            union { v16bf v; uint4 q[2]; } a;
            a.q[0] = *(const uint4*)&As[buf][row][klo];        // K = klo..klo+7
            a.q[1] = *(const uint4*)&As[buf][row][klo + 8];    // K = klo+8..klo+15
            #pragma unroll
            for (int nf = 0; nf < 2; ++nf)
                acc[mf][nf] = __builtin_amdgcn_wmma_f32_16x16x32_bf16(
                    false, a.v, false, bfr[nf], (short)0, acc[mf][nf], false, false);
        }
        __syncthreads();
        buf ^= 1;
    }

    // spill C fragments to LDS gate tile (C layout: lanes16-31 hold M+8)
    {
        const int mh = (lane >> 4) * 8;
        const int nl = lane & 15;
        #pragma unroll
        for (int mf = 0; mf < 2; ++mf)
            #pragma unroll
            for (int nf = 0; nf < 2; ++nf) {
                const int mb = waveM * 32 + mf * 16 + mh;
                const int nb = waveN * 32 + nf * 16 + nl;
                #pragma unroll
                for (int v = 0; v < 8; ++v) Gs[mb + v][nb] = acc[mf][nf][v];
            }
    }
    __syncthreads();

    // fused LSTM pointwise: gates interleaved as n = 4*j + {i,f,g,o}
    #pragma unroll
    for (int it = 0; it < 8; ++it) {
        const int p  = tid + it * 256;
        const int m  = p >> 5;
        const int jl = p & 31;
        float4 gt  = *(const float4*)&Gs[m][jl * 4];
        const int nb = n0g + jl * 4;
        float4 bxv = *(const float4*)&bx[nb];
        float4 bsv = *(const float4*)&bsum[nb];
        float gi = gt.x + tw * bxv.x + bsv.x;
        float gf = gt.y + tw * bxv.y + bsv.y;
        float gg = gt.z + tw * bxv.z + bsv.z;
        float go = gt.w + tw * bxv.w + bsv.w;
        gi = 1.f / (1.f + __expf(-gi));
        gf = 1.f / (1.f + __expf(-gf));
        gg = tanhf(gg);
        go = 1.f / (1.f + __expf(-go));
        const size_t idx = (size_t)(m0g + m) * HID + (blockIdx.y * 32 + jl);
        const float cn = gf * cbuf[idx] + gi * gg;
        cbuf[idx] = cn;
        hbuf[idx] = go * tanhf(cn);
    }
}

// ---------------- output head: sigmoid(h @ W_out.T + b_out) ----------------
__global__ void ltc_out(const float* __restrict__ hbuf, const float* __restrict__ W_out,
                        const float* __restrict__ b_out, float* __restrict__ out) {
    const int b = blockIdx.x;
    const int tid = threadIdx.x; // 256
    float p = 0.f;
    for (int k = tid; k < HID; k += 256) p += hbuf[(size_t)b * HID + k] * W_out[k];
    __shared__ float red[256];
    red[tid] = p; __syncthreads();
    for (int s = 128; s > 0; s >>= 1) { if (tid < s) red[tid] += red[tid + s]; __syncthreads(); }
    if (tid == 0) out[b] = 1.f / (1.f + __expf(-(red[0] + b_out[0])));
}

extern "C" void kernel_launch(void* const* d_in, const int* in_sizes, int n_in,
                              void* d_out, int out_size, void* d_ws, size_t ws_size,
                              hipStream_t stream) {
    const float* seq   = (const float*)d_in[0];
    const float* tiv   = (const float*)d_in[1];
    const float* W_in  = (const float*)d_in[2];
    const float* b_in  = (const float*)d_in[3];
    const float* W_ih  = (const float*)d_in[4];
    const float* W_hh  = (const float*)d_in[5];
    const float* b_ih  = (const float*)d_in[6];
    const float* b_hh  = (const float*)d_in[7];
    const float* W_out = (const float*)d_in[8];
    const float* b_out = (const float*)d_in[9];

    char* ws = (char*)d_ws;
    size_t off = 0;
    __bf16* Wpack = (__bf16*)(ws + off); off += (size_t)NG * KTOT * sizeof(__bf16); // 9 MB
    float*  bx    = (float*)(ws + off);  off += (size_t)NG * sizeof(float);
    float*  bsum  = (float*)(ws + off);  off += (size_t)NG * sizeof(float);
    float*  hbuf  = (float*)(ws + off);  off += (size_t)BATCH * HID * sizeof(float);
    float*  cbuf  = (float*)(ws + off);  off += (size_t)BATCH * HID * sizeof(float);
    (void)ws_size; (void)in_sizes; (void)n_in; (void)out_size;

    ltc_pack<<<NG, 128, 0, stream>>>(W_in, b_in, W_ih, W_hh, b_ih, b_hh, Wpack, bx, bsum);
    ltc_init<<<(BATCH * HID + 255) / 256, 256, 0, stream>>>(hbuf, cbuf);
    for (int t = 0; t < T_STEPS; ++t)
        ltc_step<<<dim3(BATCH / WG_M, NG / WG_N), 256, 0, stream>>>(
            seq, tiv, Wpack, bx, bsum, hbuf, cbuf, t);
    ltc_out<<<BATCH, 256, 0, stream>>>(hbuf, W_out, b_out, (float*)d_out);
}
